// SarcasmGNN_38431367364803
// MI455X (gfx1250) — matrix operand order repaired and checked by reading, not compiled, not run on hardware
//
#include <hip/hip_runtime.h>

// ---------------- problem constants (from reference) ----------------
constexpr int NODES  = 50000;
constexpr int EDGES  = 800000;
constexpr int DIN    = 256;
constexpr int HIDC   = 128;
constexpr int EDIM   = 32;
constexpr int NGRAPH = 64;
constexpr int EALL   = EDGES + NODES;   // edges + self loops
constexpr int JKDIM  = DIN + 2 * HIDC;  // 512

// ---------------- WMMA types ----------------
typedef __attribute__((ext_vector_type(16))) __bf16 v16bf;
typedef __attribute__((ext_vector_type(8)))  float  v8f;

// ---------------- helpers ----------------
__device__ __forceinline__ unsigned f2key(float f) {
  unsigned u = __float_as_uint(f);
  return (u & 0x80000000u) ? ~u : (u | 0x80000000u);
}
__device__ __forceinline__ float key2f(unsigned k) {
  unsigned u = (k & 0x80000000u) ? (k ^ 0x80000000u) : ~k;
  return __uint_as_float(u);
}
__device__ __forceinline__ float wave_sum(float v) {
  #pragma unroll
  for (int off = 16; off > 0; off >>= 1) v += __shfl_xor(v, off, 32);
  return v;
}
__device__ __forceinline__ float leaky02(float m) {
  return (m > 0.0f) ? m : 0.2f * m;
}

// ---------------- init kernels ----------------
__global__ void fill_f32(float* p, float v, size_t n) {
  size_t i = (size_t)blockIdx.x * blockDim.x + threadIdx.x;
  if (i < n) p[i] = v;
}
__global__ void fill_u32(unsigned* p, unsigned v, size_t n) {
  size_t i = (size_t)blockIdx.x * blockDim.x + threadIdx.x;
  if (i < n) p[i] = v;
}

// ---------------- edge feature kernel: scaled + self-loop accumulation ----
// one wave per edge, lane == feature dim (EDIM == 32 == wave32)
__global__ void edge_feat_kernel(const int* __restrict__ dst,
                                 const int* __restrict__ eattr,
                                 const float* __restrict__ ew,
                                 const float* __restrict__ irf,
                                 const float* __restrict__ eemb,
                                 float* __restrict__ scaled,
                                 float* __restrict__ loopsum,
                                 float* __restrict__ cnt) {
  int e = blockIdx.x * (blockDim.x >> 5) + (threadIdx.x >> 5);
  int lane = threadIdx.x & 31;
  if (e >= EDGES) return;
  int rel = eattr[e];
  float raw = eemb[(size_t)rel * EDIM + lane];
  float nrm = sqrtf(wave_sum(raw * raw));
  float val = raw / fmaxf(nrm, 1e-12f) * irf[rel] * ew[e];
  scaled[(size_t)e * EDIM + lane] = val;
  int d = dst[e];
  atomicAdd(&loopsum[(size_t)d * EDIM + lane], val);
  if (lane == 0) atomicAdd(&cnt[d], 1.0f);
}

__global__ void loop_feat_kernel(const float* __restrict__ loopsum,
                                 const float* __restrict__ cnt,
                                 float* __restrict__ loopfeat) {
  size_t i = (size_t)blockIdx.x * blockDim.x + threadIdx.x;
  if (i >= (size_t)NODES * EDIM) return;
  loopfeat[i] = loopsum[i] / fmaxf(cnt[i / EDIM], 1.0f);
}

// ---------------- bf16 WMMA GEMM (optionally dual-output) ------------------
// C1 = A@W1 + bias1 (and C2 = A@W2 + bias2 when DUAL), sharing one A fragment.
// 4-wave workgroup computes a 64x16 output block. B K-slices staged in LDS
// COLUMN-MAJOR (Bs[n*32+k]) so each lane's 16-K fragment is one contiguous
// 32-byte v16bf LDS read (ds_load_b128 pair). K fully unrolled.
template <int K, bool DUAL>
__global__ void gemm_wmma_bf16(const float* __restrict__ A,
                               const float* __restrict__ W1,
                               const float* __restrict__ bias1,
                               float* __restrict__ C1,
                               const float* __restrict__ W2,
                               const float* __restrict__ bias2,
                               float* __restrict__ C2,
                               int M, int Nc) {
  constexpr int KSTEPS = K / 32;
  __shared__ alignas(32) __bf16 Bs[DUAL ? 2 : 1][16 * 32];  // column-major tiles

  int tileN  = blockIdx.x * 16;
  int blockM = blockIdx.y * 64;
  int wave = threadIdx.x >> 5;     // 0..3 -> M sub-tile
  int lane = threadIdx.x & 31;
  int hl   = lane >> 4;            // wave half
  int r    = lane & 15;
  // A k order chosen here (contiguous, matches B): lanes 0-15 -> K 0..15 ; lanes 16-31 -> K 16..31
  int akb0 = hl ? 16 : 0;
  int akb1 = hl ? 24 : 8;
  // B k order chosen here (contiguous, matches A): lanes 0-15 -> K 0..15 ; lanes 16-31 -> K 16..31
  int bkb  = hl ? 16 : 0;

  int rowA = blockM + wave * 16 + r;
  if (rowA > M - 1) rowA = M - 1;              // clamp loads; stores are guarded
  const float* Arow = A + (size_t)rowA * K;

  int bk = threadIdx.x >> 2;                   // 0..31 : k within B tile
  int bn = (threadIdx.x & 3) * 4;              // 0,4,8,12 : n quad

  v8f acc1 = {};
  v8f acc2 = {};
  #pragma unroll
  for (int step = 0; step < KSTEPS; ++step) {
    const int kk = step * 32;
    __syncthreads();                           // protect prior-step LDS reads
    // cooperative B tiles: row-major global_load_b128, column-major LDS store
    {
      const float4 w4 = *(const float4*)&W1[(size_t)(kk + bk) * Nc + tileN + bn];
      Bs[0][(bn + 0) * 32 + bk] = (__bf16)w4.x;
      Bs[0][(bn + 1) * 32 + bk] = (__bf16)w4.y;
      Bs[0][(bn + 2) * 32 + bk] = (__bf16)w4.z;
      Bs[0][(bn + 3) * 32 + bk] = (__bf16)w4.w;
    }
    if constexpr (DUAL) {
      const float4 w4 = *(const float4*)&W2[(size_t)(kk + bk) * Nc + tileN + bn];
      Bs[1][(bn + 0) * 32 + bk] = (__bf16)w4.x;
      Bs[1][(bn + 1) * 32 + bk] = (__bf16)w4.y;
      Bs[1][(bn + 2) * 32 + bk] = (__bf16)w4.z;
      Bs[1][(bn + 3) * 32 + bk] = (__bf16)w4.w;
    }
    __syncthreads();

    if (step + 1 < KSTEPS) __builtin_prefetch(Arow + kk + 32, 0, 3);

    const float4 a0 = *(const float4*)(Arow + kk + akb0);
    const float4 a1 = *(const float4*)(Arow + kk + akb0 + 4);
    const float4 a2 = *(const float4*)(Arow + kk + akb1);
    const float4 a3 = *(const float4*)(Arow + kk + akb1 + 4);
    v16bf a;
    a[0] = (__bf16)a0.x; a[1] = (__bf16)a0.y; a[2] = (__bf16)a0.z; a[3] = (__bf16)a0.w;
    a[4] = (__bf16)a1.x; a[5] = (__bf16)a1.y; a[6] = (__bf16)a1.z; a[7] = (__bf16)a1.w;
    a[8] = (__bf16)a2.x; a[9] = (__bf16)a2.y; a[10] = (__bf16)a2.z; a[11] = (__bf16)a2.w;
    a[12] = (__bf16)a3.x; a[13] = (__bf16)a3.y; a[14] = (__bf16)a3.z; a[15] = (__bf16)a3.w;

    const v16bf b1 = *(const v16bf*)&Bs[0][r * 32 + bkb];  // 32B contiguous
    acc1 = __builtin_amdgcn_wmma_f32_16x16x32_bf16(false, a, false, b1,
                                                   (short)0, acc1, false, false);
    if constexpr (DUAL) {
      const v16bf b2 = *(const v16bf*)&Bs[1][r * 32 + bkb];
      acc2 = __builtin_amdgcn_wmma_f32_16x16x32_bf16(false, a, false, b2,
                                                     (short)0, acc2, false, false);
    }
  }

  // C/D layout: lane holds column (lane&15); rows hl*8 + 0..7 in acc[0..7]
  int col = tileN + r;
  float bv1 = bias1[col];
  float bv2 = DUAL ? bias2[col] : 0.0f;
  #pragma unroll
  for (int i = 0; i < 8; ++i) {
    int row = blockM + wave * 16 + hl * 8 + i;
    if (row < M) {
      C1[(size_t)row * Nc + col] = acc1[i] + bv1;
      if constexpr (DUAL) C2[(size_t)row * Nc + col] = acc2[i] + bv2;
    }
  }
}

// ---------------- attention pass 1: score + segment max --------------------
// one wave per (real or loop) edge; lane owns channels 4*lane..4*lane+3 so
// every gather is a single b128 load; fuses ef = feat @ We on the fly
template <int H>
__global__ void edge_score_kernel(const int* __restrict__ src,
                                  const int* __restrict__ dst,
                                  const float* __restrict__ scaled,
                                  const float* __restrict__ loopfeat,
                                  const float* __restrict__ XL,
                                  const float* __restrict__ XR,
                                  const float* __restrict__ We,   // [32, H*128]
                                  const float* __restrict__ att,  // [H, 128]
                                  float* __restrict__ score,      // [EALL, H]
                                  unsigned* __restrict__ mxkey) { // [NODES, H]
  int e = blockIdx.x * (blockDim.x >> 5) + (threadIdx.x >> 5);
  int lane = threadIdx.x & 31;
  if (e >= EALL) return;
  int s, d;
  const float* feat;
  if (e < EDGES) { s = src[e]; d = dst[e]; feat = scaled + (size_t)e * EDIM; }
  else           { s = d = e - EDGES;      feat = loopfeat + (size_t)(e - EDGES) * EDIM; }

  float fk[EDIM];
  const float4* f4 = (const float4*)feat;
  #pragma unroll
  for (int k = 0; k < EDIM / 4; ++k) {
    float4 v = f4[k];
    fk[4 * k] = v.x; fk[4 * k + 1] = v.y; fk[4 * k + 2] = v.z; fk[4 * k + 3] = v.w;
  }

  int c4 = lane * 4;
  #pragma unroll
  for (int h = 0; h < H; ++h) {
    const float4 xl = *(const float4*)&XL[((size_t)s * H + h) * HIDC + c4];
    const float4 xr = *(const float4*)&XR[((size_t)d * H + h) * HIDC + c4];
    float ex_ = 0.0f, ey = 0.0f, ez = 0.0f, ew_ = 0.0f;
    #pragma unroll
    for (int k = 0; k < EDIM; ++k) {
      const float4 w4 = *(const float4*)&We[(size_t)k * (H * HIDC) + h * HIDC + c4];
      ex_ += fk[k] * w4.x; ey += fk[k] * w4.y; ez += fk[k] * w4.z; ew_ += fk[k] * w4.w;
    }
    const float4 at = *(const float4*)&att[h * HIDC + c4];
    float partial = leaky02(xl.x + xr.x + ex_) * at.x
                  + leaky02(xl.y + xr.y + ey)  * at.y
                  + leaky02(xl.z + xr.z + ez)  * at.z
                  + leaky02(xl.w + xr.w + ew_) * at.w;
    float sc = wave_sum(partial);
    if (lane == 0) {
      score[(size_t)e * H + h] = sc;
      atomicMax(&mxkey[(size_t)d * H + h], f2key(sc));
    }
  }
}

// ---------------- attention pass 2: exp + segment sum ----------------------
__global__ void edge_softmax_kernel(const int* __restrict__ dst, int H,
                                    const unsigned* __restrict__ mxkey,
                                    float* __restrict__ score,
                                    float* __restrict__ den) {
  size_t i = (size_t)blockIdx.x * blockDim.x + threadIdx.x;
  if (i >= (size_t)EALL * H) return;
  int e = (int)(i / H), h = (int)(i - (size_t)e * H);
  int d = (e < EDGES) ? dst[e] : (e - EDGES);
  float m = key2f(mxkey[(size_t)d * H + h]);
  if (!(fabsf(m) < 3.0e38f)) m = 0.0f;  // where(isfinite(mx), mx, 0)
  float ex = __expf(score[i] - m);
  atomicAdd(&den[(size_t)d * H + h], ex);
  score[i] = ex;
}

// ---------------- attention pass 3: weighted scatter -----------------------
template <int H>
__global__ void edge_aggregate_kernel(const int* __restrict__ src,
                                      const int* __restrict__ dst,
                                      const float* __restrict__ ex,
                                      const float* __restrict__ den,
                                      const float* __restrict__ XL,
                                      float* __restrict__ outb) {
  int e = blockIdx.x * (blockDim.x >> 5) + (threadIdx.x >> 5);
  int lane = threadIdx.x & 31;
  if (e >= EALL) return;
  int s, d;
  if (e < EDGES) { s = src[e]; d = dst[e]; } else { s = d = e - EDGES; }
  int c4 = lane * 4;
  #pragma unroll
  for (int h = 0; h < H; ++h) {
    float alpha = ex[(size_t)e * H + h] / (den[(size_t)d * H + h] + 1e-16f);
    const float4 xl = *(const float4*)&XL[((size_t)s * H + h) * HIDC + c4];
    float* op = &outb[((size_t)d * H + h) * HIDC + c4];
    atomicAdd(op + 0, alpha * xl.x);
    atomicAdd(op + 1, alpha * xl.y);
    atomicAdd(op + 2, alpha * xl.z);
    atomicAdd(op + 3, alpha * xl.w);
  }
}

// ---------------- layer combiners ----------------
__global__ void combine1_kernel(const float* __restrict__ out1,  // [N,2,128]
                                const float* __restrict__ bias1,
                                const float* __restrict__ xskip,
                                float* __restrict__ h1) {
  size_t i = (size_t)blockIdx.x * blockDim.x + threadIdx.x;
  if (i >= (size_t)NODES * HIDC) return;
  size_t n = i / HIDC; int c = (int)(i - n * HIDC);
  float v = 0.5f * (out1[n * 2 * HIDC + c] + out1[n * 2 * HIDC + HIDC + c]);
  v += bias1[c] + xskip[i];
  h1[i] = fmaxf(v, 0.0f);
}

__global__ void combine2_kernel(const float* __restrict__ out2,  // [N,128]
                                const float* __restrict__ bias2,
                                float* __restrict__ h2) {
  size_t i = (size_t)blockIdx.x * blockDim.x + threadIdx.x;
  if (i >= (size_t)NODES * HIDC) return;
  int c = (int)(i % HIDC);
  h2[i] = fmaxf(out2[i] + bias2[c], 0.0f);
}

// ---------------- graph pooling ----------------
__global__ void graph_count_kernel(const int* __restrict__ batch, float* __restrict__ gcnt) {
  int i = blockIdx.x * blockDim.x + threadIdx.x;
  if (i < NODES) atomicAdd(&gcnt[batch[i]], 1.0f);
}

__global__ void pool_kernel(const int* __restrict__ batch,
                            const float* __restrict__ x,
                            const float* __restrict__ h1,
                            const float* __restrict__ h2,
                            unsigned* __restrict__ pmax,
                            float* __restrict__ psum) {
  size_t i = (size_t)blockIdx.x * blockDim.x + threadIdx.x;
  if (i >= (size_t)NODES * JKDIM) return;
  size_t n = i / JKDIM; int c = (int)(i - n * JKDIM);
  int g = batch[n];
  float v;
  if (c < DIN)              v = x[n * DIN + c];
  else if (c < DIN + HIDC)  v = h1[n * HIDC + (c - DIN)];
  else                      v = h2[n * HIDC + (c - DIN - HIDC)];
  atomicMax(&pmax[(size_t)g * JKDIM + c], f2key(v));
  atomicAdd(&psum[(size_t)g * JKDIM + c], v);
}

// ---------------- readout MLP ----------------
__global__ void mlp1_kernel(const unsigned* __restrict__ pmax,
                            const float* __restrict__ psum,
                            const float* __restrict__ gcnt,
                            const float* __restrict__ W1,
                            const float* __restrict__ b1,
                            float* __restrict__ h) {
  int g = blockIdx.x, c = threadIdx.x;  // 64 x 128
  float acc = b1[c];
  float inv = 1.0f / fmaxf(gcnt[g], 1.0f);
  for (int k = 0; k < JKDIM; ++k) {
    float v = key2f(pmax[(size_t)g * JKDIM + k]);
    if (!(fabsf(v) < 3.0e38f)) v = 0.0f;
    acc += v * W1[(size_t)k * HIDC + c];
  }
  for (int k = 0; k < JKDIM; ++k) {
    acc += (psum[(size_t)g * JKDIM + k] * inv) * W1[(size_t)(JKDIM + k) * HIDC + c];
  }
  h[(size_t)g * HIDC + c] = fmaxf(acc, 0.0f);
}

__global__ void mlp2_kernel(const float* __restrict__ h,
                            const float* __restrict__ W2,
                            const float* __restrict__ b2,
                            float* __restrict__ outv) {
  int g = threadIdx.x;
  if (g >= NGRAPH) return;
  float acc = b2[0];
  for (int c = 0; c < HIDC; ++c) acc += h[(size_t)g * HIDC + c] * W2[c];
  outv[g] = acc;
}

// ---------------- launch ----------------
extern "C" void kernel_launch(void* const* d_in, const int* in_sizes, int n_in,
                              void* d_out, int out_size, void* d_ws, size_t ws_size,
                              hipStream_t stream) {
  (void)in_sizes; (void)n_in; (void)out_size; (void)ws_size;
  const float* x     = (const float*)d_in[0];
  const int*   ei    = (const int*)  d_in[1];
  const int*   eattr = (const int*)  d_in[2];
  const float* ew    = (const float*)d_in[3];
  const int*   batch = (const int*)  d_in[4];
  const float* irf   = (const float*)d_in[5];
  const float* eemb  = (const float*)d_in[6];
  const float* Wl1   = (const float*)d_in[7];
  const float* bl1   = (const float*)d_in[8];
  const float* Wr1   = (const float*)d_in[9];
  const float* br1   = (const float*)d_in[10];
  const float* We1   = (const float*)d_in[11];
  const float* att1  = (const float*)d_in[12];
  const float* bias1 = (const float*)d_in[13];
  const float* Wl2   = (const float*)d_in[14];
  const float* bl2   = (const float*)d_in[15];
  const float* Wr2   = (const float*)d_in[16];
  const float* br2   = (const float*)d_in[17];
  const float* We2   = (const float*)d_in[18];
  const float* att2  = (const float*)d_in[19];
  const float* bias2 = (const float*)d_in[20];
  const float* skipW = (const float*)d_in[21];
  const float* skipb = (const float*)d_in[22];
  const float* W1    = (const float*)d_in[23];
  const float* b1    = (const float*)d_in[24];
  const float* W2    = (const float*)d_in[25];
  const float* b2    = (const float*)d_in[26];

  const int* src = ei;            // edge_index[0]
  const int* dst = ei + EDGES;    // edge_index[1]

  float* outv = (float*)d_out;          // [64]
  float* h2   = outv + NGRAPH;          // [N,128] second tuple element

  // ---- workspace layout (floats) ----
  float* ws = (float*)d_ws;
  size_t o = 0;
  float*    scaled   = ws + o; o += (size_t)EDGES * EDIM;   // 25.6M
  float*    loopsum  = ws + o; o += (size_t)NODES * EDIM;
  float*    loopfeat = ws + o; o += (size_t)NODES * EDIM;
  float*    cnt      = ws + o; o += NODES;
  float*    XL1      = ws + o; o += (size_t)NODES * 256;    // also XL2|XR2 later
  float*    XR1      = ws + o; o += (size_t)NODES * 256;    // also out1 later
  float*    xskip    = ws + o; o += (size_t)NODES * HIDC;
  float*    score1   = ws + o; o += (size_t)EALL * 2;
  unsigned* mx1      = (unsigned*)(ws + o); o += (size_t)NODES * 2;
  float*    den1     = ws + o; o += (size_t)NODES * 2;
  float*    h1       = ws + o; o += (size_t)NODES * HIDC;
  float*    score2   = ws + o; o += (size_t)EALL;
  unsigned* mx2      = (unsigned*)(ws + o); o += NODES;
  float*    den2     = ws + o; o += NODES;
  unsigned* pmax     = (unsigned*)(ws + o); o += (size_t)NGRAPH * JKDIM;
  float*    psum     = ws + o; o += (size_t)NGRAPH * JKDIM;
  float*    gcnt     = ws + o; o += NGRAPH;
  float*    mlph     = ws + o; o += (size_t)NGRAPH * HIDC;
  // aliases (lifetimes do not overlap):
  float* out1 = XR1;              // [N,2,128] zeroed after pass1-L1 (XR1 dead)
  float* XL2  = XL1;              // [N,128]   after pass3-L1 (XL1 dead)
  float* XR2  = XL1 + (size_t)NODES * HIDC;
  float* out2 = scaled;           // [N,128]   after pass1-L2 (scaled dead)

  auto blocksF = [](size_t n) { return (unsigned)((n + 255) / 256); };
  const int EB = (EALL + 7) / 8;        // 8 waves (edges) per 256-thread block
  const int EB_real = (EDGES + 7) / 8;
  const unsigned GY = (NODES + 63) / 64;  // 64-row GEMM blocks

  // 1) edge features + self-loop means
  fill_f32<<<blocksF((size_t)NODES * EDIM), 256, 0, stream>>>(loopsum, 0.0f, (size_t)NODES * EDIM);
  fill_f32<<<blocksF(NODES), 256, 0, stream>>>(cnt, 0.0f, NODES);
  edge_feat_kernel<<<EB_real, 256, 0, stream>>>(dst, eattr, ew, irf, eemb, scaled, loopsum, cnt);
  loop_feat_kernel<<<blocksF((size_t)NODES * EDIM), 256, 0, stream>>>(loopsum, cnt, loopfeat);

  // 2) dense projections via bf16 WMMA; Wl/Wr fused to read A once
  gemm_wmma_bf16<256, true><<<dim3(16, GY), 128, 0, stream>>>(
      x, Wl1, bl1, XL1, Wr1, br1, XR1, NODES, 256);
  gemm_wmma_bf16<256, false><<<dim3(8, GY), 128, 0, stream>>>(
      x, skipW, skipb, xskip, nullptr, nullptr, nullptr, NODES, 128);

  // 3) layer-1 attention (H=2, C=128)
  fill_u32<<<blocksF((size_t)NODES * 2), 256, 0, stream>>>(mx1, 0u, (size_t)NODES * 2);
  fill_f32<<<blocksF((size_t)NODES * 2), 256, 0, stream>>>(den1, 0.0f, (size_t)NODES * 2);
  edge_score_kernel<2><<<EB, 256, 0, stream>>>(src, dst, scaled, loopfeat,
                                               XL1, XR1, We1, att1, score1, mx1);
  edge_softmax_kernel<<<blocksF((size_t)EALL * 2), 256, 0, stream>>>(dst, 2, mx1, score1, den1);
  fill_f32<<<blocksF((size_t)NODES * 256), 256, 0, stream>>>(out1, 0.0f, (size_t)NODES * 256);
  edge_aggregate_kernel<2><<<EB, 256, 0, stream>>>(src, dst, score1, den1, XL1, out1);
  combine1_kernel<<<blocksF((size_t)NODES * HIDC), 256, 0, stream>>>(out1, bias1, xskip, h1);

  // 4) layer-2 projections + attention (H=1, C=128)
  gemm_wmma_bf16<128, true><<<dim3(8, GY), 128, 0, stream>>>(
      h1, Wl2, bl2, XL2, Wr2, br2, XR2, NODES, 128);
  fill_u32<<<blocksF(NODES), 256, 0, stream>>>(mx2, 0u, NODES);
  fill_f32<<<blocksF(NODES), 256, 0, stream>>>(den2, 0.0f, NODES);
  edge_score_kernel<1><<<EB, 256, 0, stream>>>(src, dst, scaled, loopfeat,
                                               XL2, XR2, We2, att2, score2, mx2);
  edge_softmax_kernel<<<blocksF(EALL), 256, 0, stream>>>(dst, 1, mx2, score2, den2);
  fill_f32<<<blocksF((size_t)NODES * HIDC), 256, 0, stream>>>(out2, 0.0f, (size_t)NODES * HIDC);
  edge_aggregate_kernel<1><<<EB, 256, 0, stream>>>(src, dst, score2, den2, XL2, out2);
  combine2_kernel<<<blocksF((size_t)NODES * HIDC), 256, 0, stream>>>(out2, bias2, h2);

  // 5) JK pooling per graph (max + mean over [x | h1 | h2])
  fill_u32<<<blocksF((size_t)NGRAPH * JKDIM), 256, 0, stream>>>(pmax, 0u, (size_t)NGRAPH * JKDIM);
  fill_f32<<<blocksF((size_t)NGRAPH * JKDIM), 256, 0, stream>>>(psum, 0.0f, (size_t)NGRAPH * JKDIM);
  fill_f32<<<1, 256, 0, stream>>>(gcnt, 0.0f, NGRAPH);
  graph_count_kernel<<<blocksF(NODES), 256, 0, stream>>>(batch, gcnt);
  pool_kernel<<<blocksF((size_t)NODES * JKDIM), 256, 0, stream>>>(batch, x, h1, h2, pmax, psum);

  // 6) readout MLP
  mlp1_kernel<<<NGRAPH, HIDC, 0, stream>>>(pmax, psum, gcnt, W1, b1, mlph);
  mlp2_kernel<<<1, 64, 0, stream>>>(mlph, W2, b2, outv);
}
